// AFM_31267361915374
// MI455X (gfx1250) — matrix operand order repaired and checked by reading, not compiled, not run on hardware
//
#include <hip/hip_runtime.h>
#include <hip/hip_bf16.h>
#include <math.h>

// CDNA5 (gfx1250) wave32 WMMA types
typedef __attribute__((ext_vector_type(16))) __bf16 v16bf;
typedef __attribute__((ext_vector_type(8)))  __bf16 v8bf;
typedef __attribute__((ext_vector_type(8)))  float  v8f;

#define F_FIELDS  39
#define NPAIR     741      // 39*38/2
#define NPAIR_PAD 752      // 47 * 16
#define EMB       64
#define ATT       64
#define NTILE     47       // ceil(741/16)
#define NTHREADS  128      // 4 wave32s: each wave owns a 16-wide N-slice of ATT

// VALU-only DPP16 row reduction step: v += perm(v). No DS traffic, no dscnt waits.
#define DPP_ADD_F32(v, ctrl)                                                   \
    do {                                                                       \
        int _t = __builtin_amdgcn_update_dpp(                                  \
            0, __builtin_bit_cast(int, (v)), (ctrl), 0xf, 0xf, true);          \
        (v) += __builtin_bit_cast(float, _t);                                  \
    } while (0)

__global__ __launch_bounds__(NTHREADS) void afm_fused_kernel(
    const int*   __restrict__ feat_index,   // [B,39] int32
    const float* __restrict__ feat_value,   // [B,39]
    const float* __restrict__ fo_w,         // [100000,1]
    const float* __restrict__ emb_table,    // [100000,64]
    const float* __restrict__ att_W,        // [64,64]  (d, a)
    const float* __restrict__ att_b,        // [64]
    const float* __restrict__ att_h,        // [64]
    const float* __restrict__ p_vec,        // [64]
    const float* __restrict__ bias,         // [1]
    float*       __restrict__ out)          // [B]
{
    __shared__ int    fi[F_FIELDS];
    __shared__ float  fvs[F_FIELDS];
    __shared__ __align__(16) float  emb_s[F_FIELDS * EMB];   // scaled embeddings, f32
    __shared__ __align__(16) __bf16 wt[ATT * EMB];           // wt[a*64+d] = bf16(att_W[d][a])
    __shared__ float  attb[ATT], atth[ATT], pvec[EMB];
    __shared__ unsigned short pairT[NPAIR_PAD];              // row | col<<8
    __shared__ __align__(16) __bf16 interT[16 * EMB];        // current 16-pair tile, bf16
    __shared__ float  sig[NPAIR_PAD];
    __shared__ float  pooled[EMB];
    __shared__ float  red[NTHREADS];
    __shared__ float  s_yfirst, s_max, s_sum;

    const int tid = threadIdx.x;
    const int b   = blockIdx.x;

    // ---- Stage A: indices/values ----
    if (tid < F_FIELDS) {
        fi[tid]  = feat_index[b * F_FIELDS + tid];
        fvs[tid] = feat_value[b * F_FIELDS + tid];
    }
    __syncthreads();

    // Gather scaled embeddings (emb_table is L2-resident: 25.6MB << 192MB L2)
    for (int idx = tid; idx < F_FIELDS * EMB; idx += NTHREADS) {
        int f = idx >> 6, d = idx & 63;
        emb_s[idx] = emb_table[fi[f] * EMB + d] * fvs[f];
    }
    // att_W transposed -> bf16 LDS (contiguous K per output column)
    for (int idx = tid; idx < ATT * EMB; idx += NTHREADS) {
        int a = idx >> 6, d = idx & 63;
        wt[idx] = (__bf16)att_W[d * ATT + a];
    }
    if (tid < ATT) { attb[tid] = att_b[tid]; atth[tid] = att_h[tid]; pvec[tid] = p_vec[tid]; }
    if (tid < EMB) pooled[tid] = 0.f;

    // Pair table (np.triu_indices(39,k=1) row-major) packed u16, zero sig
    for (int p = tid; p < NPAIR_PAD; p += NTHREADS) {
        sig[p] = 0.f;
        unsigned short pt = 0;
        if (p < NPAIR) {
            int pp = p, i = 0, rem = F_FIELDS - 1;
            while (pp >= rem) { pp -= rem; --rem; ++i; }
            pt = (unsigned short)(i | ((i + 1 + pp) << 8));
        }
        pairT[p] = pt;
    }

    // y_first = sum_f fo_w[idx]*val (tree reduction)
    red[tid] = (tid < F_FIELDS) ? fo_w[fi[tid]] * fvs[tid] : 0.f;
    __syncthreads();
    for (int off = NTHREADS / 2; off > 0; off >>= 1) {
        if (tid < off) red[tid] += red[tid + off];
        __syncthreads();
    }
    if (tid == 0) s_yfirst = red[0];
    __syncthreads();

    // ---- Stage B: attention logits via bf16 WMMA ----
    const int lane  = tid & 31;
    const int wv    = tid >> 5;               // wave id: N-slice [16*wv, 16*wv+16)
    const int mhalf = (lane < 16) ? 0 : 8;    // C-layout M offset for upper half-wave
    const int halfA = (lane < 16) ? 0 : 8;    // 16-bit A 16x32 lane K-offset
    const int halfB = (lane < 16) ? 0 : 16;   // unused: B fragments use halfA, same K order as A
    const int ncol  = wv * 16 + (lane & 15);  // output column (att dim)
    const float ab_n = attb[ncol];
    const float ah_n = atth[ncol];

    // Hoist loop-invariant B fragments (att_W) into registers: saves 4 ds_load_b128/tile
    v16bf bv0, bv1;
#pragma unroll
    for (int i = 0; i < 16; ++i) {
        bv0[i] = wt[ncol * EMB +      halfA + (i & 7) + ((i >> 3) << 4)];   // K chunk 0: K = halfA+i (i<8), 16+halfA+i-8 (i>=8)
        bv1[i] = wt[ncol * EMB + 32 + halfA + (i & 7) + ((i >> 3) << 4)];   // K chunk 1
    }

    // interT build mapping: thread owns row m = tid>>3, contiguous d span (tid&7)*8
    const int bm = tid >> 3;
    const int bd = (tid & 7) * 8;
    const int mrow = lane & 15;               // A-matrix row owned by this lane

    for (int T = 0; T < NTILE; ++T) {
        // Build 16x64 bf16 inter tile: 4x ds_load_b128 + 1x ds_store_b128 per thread
        {
            v8bf st;
            const int p = T * 16 + bm;
            if (p < NPAIR) {
                unsigned short pt = pairT[p];
                const float* er = &emb_s[(pt & 255) * EMB + bd];
                const float* ec = &emb_s[(pt >> 8)  * EMB + bd];
#pragma unroll
                for (int i = 0; i < 8; ++i)
                    st[i] = (__bf16)(er[i] * ec[i]);
            } else {
#pragma unroll
                for (int i = 0; i < 8; ++i) st[i] = (__bf16)0.f;
            }
            *(v8bf*)&interT[bm * EMB + bd] = st;
        }
        __syncthreads();

        // D = inter_tile(16x64) @ att_W(64x16-slice): two K=32 bf16 WMMAs, f32 acc
        v16bf av0, av1;
#pragma unroll
        for (int i = 0; i < 8; ++i) {
            av0[i]     = interT[mrow * EMB +      halfA + i];
            av0[8 + i] = interT[mrow * EMB + 16 + halfA + i];
            av1[i]     = interT[mrow * EMB + 32 + halfA + i];
            av1[8 + i] = interT[mrow * EMB + 48 + halfA + i];
        }
        v8f acc = {};
        acc = __builtin_amdgcn_wmma_f32_16x16x32_bf16(
                  false, av0, false, bv0, (short)0, acc, false, false);
        acc = __builtin_amdgcn_wmma_f32_16x16x32_bf16(
                  false, av1, false, bv1, (short)0, acc, false, false);

        // relu(x + b) * h, then VALU-only DPP16 reduction over the 16 N-lanes.
        // quad_perm xor-1/xor-2 form quad sums; row_shr:4 + row_shr:8 accumulate
        // quads upward: full row sum lands in lanes 15 and 31.
#pragma unroll
        for (int r = 0; r < 8; ++r) {
            float s = fmaxf(acc[r] + ab_n, 0.f) * ah_n;
            DPP_ADD_F32(s, 0xB1);   // quad_perm [1,0,3,2]
            DPP_ADD_F32(s, 0x4E);   // quad_perm [2,3,0,1]
            DPP_ADD_F32(s, 0x114);  // row_shr:4
            DPP_ADD_F32(s, 0x118);  // row_shr:8
            if ((lane & 15) == 15) {
                int p = T * 16 + r + mhalf;   // C layout: VGPR r -> M = r (+8 upper half)
                if (p < NPAIR) atomicAdd(&sig[p], s);
            }
        }
        __syncthreads();   // interT rewritten next iteration
    }

    // ---- Stage C: softmax over 741 pairs ----
    float mx = -3.402823466e38f;
    for (int p = tid; p < NPAIR; p += NTHREADS) mx = fmaxf(mx, sig[p]);
    red[tid] = mx;
    __syncthreads();
    for (int off = NTHREADS / 2; off > 0; off >>= 1) {
        if (tid < off) red[tid] = fmaxf(red[tid], red[tid + off]);
        __syncthreads();
    }
    if (tid == 0) s_max = red[0];
    __syncthreads();
    mx = s_max;

    float sm = 0.f;
    for (int p = tid; p < NPAIR; p += NTHREADS) {
        float e = __expf(sig[p] - mx);
        sig[p] = e;
        sm += e;
    }
    red[tid] = sm;
    __syncthreads();
    for (int off = NTHREADS / 2; off > 0; off >>= 1) {
        if (tid < off) red[tid] += red[tid + off];
        __syncthreads();
    }
    if (tid == 0) s_sum = red[0];
    __syncthreads();

    // ---- attention pooling: pooled[d] = sum_p e_p * inter[p][d] ----
    // inter recomputed on the fly (avoids 388MB HBM materialization).
    // Each thread: float4 of d, 8-way p split -> 3 DS loads per 4 outputs/iter.
    {
        const int dg = (tid & 15) * 4;   // d group
        const int sp = tid >> 4;         // p split 0..7
        float ax = 0.f, ay = 0.f, az = 0.f, aw = 0.f;
        for (int p = sp; p < NPAIR; p += 8) {
            unsigned short pt = pairT[p];
            const float4 er = *(const float4*)&emb_s[(pt & 255) * EMB + dg];
            const float4 ec = *(const float4*)&emb_s[(pt >> 8)  * EMB + dg];
            const float e = sig[p];
            ax += e * er.x * ec.x;
            ay += e * er.y * ec.y;
            az += e * er.z * ec.z;
            aw += e * er.w * ec.w;
        }
        atomicAdd(&pooled[dg + 0], ax);
        atomicAdd(&pooled[dg + 1], ay);
        atomicAdd(&pooled[dg + 2], az);
        atomicAdd(&pooled[dg + 3], aw);
    }
    __syncthreads();

    if (tid < EMB) red[tid] = pooled[tid] * pvec[tid];
    __syncthreads();
    for (int off = 32; off > 0; off >>= 1) {
        if (tid < off) red[tid] += red[tid + off];
        __syncthreads();
    }
    if (tid == 0) {
        float att_pool = red[0] / s_sum;
        float z = bias[0] + s_yfirst + att_pool;
        out[b] = 1.f / (1.f + __expf(-z));
    }
}

extern "C" void kernel_launch(void* const* d_in, const int* in_sizes, int n_in,
                              void* d_out, int out_size, void* d_ws, size_t ws_size,
                              hipStream_t stream) {
    const int*   feat_index = (const int*)  d_in[0];
    const float* feat_value = (const float*)d_in[1];
    const float* fo_w       = (const float*)d_in[2];
    const float* emb_table  = (const float*)d_in[3];
    const float* att_W      = (const float*)d_in[4];
    const float* att_b      = (const float*)d_in[5];
    const float* att_h      = (const float*)d_in[6];
    const float* p_vec      = (const float*)d_in[7];
    const float* bias       = (const float*)d_in[8];
    float* out = (float*)d_out;

    const int B = in_sizes[0] / F_FIELDS;   // 2048
    afm_fused_kernel<<<B, NTHREADS, 0, stream>>>(
        feat_index, feat_value, fo_w, emb_table,
        att_W, att_b, att_h, p_vec, bias, out);
}
